// MHA_48223892799798
// MI455X (gfx1250) — matrix operand order repaired and checked by reading, not compiled, not run on hardware
//
#include <hip/hip_runtime.h>
#include <hip/hip_bf16.h>

typedef __bf16 bf16_t;
typedef __attribute__((ext_vector_type(16))) __bf16 v16bf;
typedef __attribute__((ext_vector_type(8)))  float  v8f;
typedef int v4i_vs __attribute__((vector_size(16)));   // matches builtin param type

union FragAB { v16bf v; uint4 q[2]; };

__device__ __forceinline__ v8f wmma_bf16(const FragAB& a, const FragAB& b, v8f c) {
    return __builtin_amdgcn_wmma_f32_16x16x32_bf16(false, a.v, false, b.v,
                                                   (short)0, c, false, false);
}

// ---- async global->LDS copy (CDNA5 GLOBAL_LOAD_ASYNC_TO_LDS_B128), guarded fallback
#if defined(__has_builtin)
#  if __has_builtin(__builtin_amdgcn_global_load_async_to_lds_b128)
#    define HAVE_ASYNC_COPY 1
#  endif
#endif
#ifndef HAVE_ASYNC_COPY
#  define HAVE_ASYNC_COPY 0
#endif

__device__ __forceinline__ void copy16_async(const bf16_t* g, bf16_t* l) {
#if HAVE_ASYNC_COPY
    __builtin_amdgcn_global_load_async_to_lds_b128(
        (__attribute__((address_space(1))) v4i_vs*)g,
        (__attribute__((address_space(3))) v4i_vs*)l, 0, 0);
#else
    *reinterpret_cast<uint4*>(l) = *reinterpret_cast<const uint4*>(g);
#endif
}

__device__ __forceinline__ void async_wait0() {
#if HAVE_ASYNC_COPY
#  if __has_builtin(__builtin_amdgcn_s_wait_asynccnt)
    __builtin_amdgcn_s_wait_asynccnt(0);
#  else
    asm volatile("s_wait_asynccnt 0x0" ::: "memory");
#  endif
#endif
}

__device__ __forceinline__ float rcp_fast(float x) {
#if defined(__has_builtin) && __has_builtin(__builtin_amdgcn_rcpf)
    return __builtin_amdgcn_rcpf(x);
#else
    return 1.0f / x;
#endif
}

__device__ __forceinline__ float exp2_fast(float x) {
#if defined(__has_builtin) && __has_builtin(__builtin_amdgcn_exp2f)
    return __builtin_amdgcn_exp2f(x);     // raw v_exp_f32 (hardware exp2)
#else
    return exp2f(x);
#endif
}

// ---------------------------------------------------------------- conversion
__global__ __launch_bounds__(256) void f32_to_bf16_k(const float* __restrict__ in,
                                                     bf16_t* __restrict__ out, int n) {
    int i = blockIdx.x * 256 + threadIdx.x;
    if (i < n) out[i] = (bf16_t)in[i];
}

// 512x512 transpose + convert: out[n][k] = (bf16) in[k][n]
__global__ __launch_bounds__(256) void transpose_f32_to_bf16_k(const float* __restrict__ in,
                                                               bf16_t* __restrict__ out) {
    __shared__ float t[32][33];
    const int tx = threadIdx.x & 31, ty = threadIdx.x >> 5;    // 32 x 8
    const int x = blockIdx.x * 32 + tx;
    const int y0 = blockIdx.y * 32;
    #pragma unroll
    for (int r = 0; r < 4; ++r)
        t[ty + r * 8][tx] = in[(size_t)(y0 + ty + r * 8) * 512 + x];
    __syncthreads();
    #pragma unroll
    for (int r = 0; r < 4; ++r)
        out[(size_t)(blockIdx.x * 32 + ty + r * 8) * 512 + y0 + tx] = (bf16_t)t[tx][ty + r * 8];
}

// ---------------------------------------------------------------- GEMM  out = (A[M,512] @ W[512,512] + bias) * scale
// W supplied PRE-TRANSPOSED (WT[n][k]) so both tiles stage with pure async b128.
// Block tile 128x128, K-step 32, double-buffered LDS.
// 8 waves in 4(M) x 2(N): each wave computes 32x64 = 8 WMMA accumulators.
// MODE: 0 = f32 row-major, 1 = bf16 row-major, 2 = bf16 scattered to [b,h,d,k] (V-transposed)
#define BM 128
#define BN 128
#define BK 32

template <int MODE>
__global__ __launch_bounds__(256) void gemm512_k(const bf16_t* __restrict__ A,
                                                 const bf16_t* __restrict__ WT,
                                                 const float*  __restrict__ bias,
                                                 bf16_t* __restrict__ outb,
                                                 float*  __restrict__ outf,
                                                 float scale) {
    __shared__ __align__(16) bf16_t As[2][BM][BK + 8];    // 2 x 128 x 40
    __shared__ __align__(16) bf16_t BsT[2][BN][BK + 8];   // 2 x 128 x 40 ([n][k])

    const int tid  = threadIdx.x;
    const int lane = tid & 31;
    const int wave = tid >> 5;
    const int hi   = lane >> 4;
    const int l16  = lane & 15;
    const int wm   = wave >> 1;          // 0..3
    const int wn   = wave & 1;           // 0..1
    const int m0   = blockIdx.y * BM;
    const int n0   = blockIdx.x * BN;

    v8f acc[2][4] = {};

    auto stage = [&](int buf, int kb) {
        #pragma unroll
        for (int it = 0; it < 2; ++it) {
            int c = tid + it * 256;               // 0..511
            int r = c >> 2, cc = (c & 3) * 8;
            copy16_async(A  + (size_t)(m0 + r) * 512 + kb + cc, &As[buf][r][cc]);
            copy16_async(WT + (size_t)(n0 + r) * 512 + kb + cc, &BsT[buf][r][cc]);
        }
    };

    stage(0, 0);
    async_wait0();
    __syncthreads();

    int buf = 0;
    for (int kb = 0; kb < 512; kb += BK, buf ^= 1) {
        if (kb + BK < 512) stage(buf ^ 1, kb + BK);   // overlap next fetch with WMMAs

        FragAB fa0, fa1;
        fa0.q[0] = *reinterpret_cast<const uint4*>(&As[buf][wm * 32 + l16][hi * 8]);
        fa0.q[1] = *reinterpret_cast<const uint4*>(&As[buf][wm * 32 + l16][16 + hi * 8]);
        fa1.q[0] = *reinterpret_cast<const uint4*>(&As[buf][wm * 32 + 16 + l16][hi * 8]);
        fa1.q[1] = *reinterpret_cast<const uint4*>(&As[buf][wm * 32 + 16 + l16][16 + hi * 8]);

        #pragma unroll
        for (int t = 0; t < 4; ++t) {
            FragAB fb;
            fb.q[0] = *reinterpret_cast<const uint4*>(&BsT[buf][wn * 64 + t * 16 + l16][hi * 8]);
            fb.q[1] = *reinterpret_cast<const uint4*>(&BsT[buf][wn * 64 + t * 16 + l16][16 + hi * 8]);
            acc[0][t] = wmma_bf16(fa0, fb, acc[0][t]);
            acc[1][t] = wmma_bf16(fa1, fb, acc[1][t]);
        }
        async_wait0();
        __syncthreads();
    }

    #pragma unroll
    for (int mi = 0; mi < 2; ++mi)
        #pragma unroll
        for (int t = 0; t < 4; ++t) {
            int col = n0 + wn * 64 + t * 16 + l16;
            float bv = bias[col];
            #pragma unroll
            for (int i = 0; i < 8; ++i) {
                int row = m0 + wm * 32 + mi * 16 + i + 8 * hi;
                float v = (acc[mi][t][i] + bv) * scale;
                if (MODE == 0) {
                    outf[(size_t)row * 512 + col] = v;
                } else if (MODE == 1) {
                    outb[(size_t)row * 512 + col] = (bf16_t)v;
                } else {
                    // V-transposed: row = b*2048+k, col = h*64+d -> [ (b*8+h)*64+d ][ k ]
                    int bI = row >> 11, k = row & 2047;
                    outb[(size_t)(bI * 512 + col) * 2048 + k] = (bf16_t)v;
                }
            }
        }
}

// ---------------------------------------------------------------- flash attention
// grid: (L/128, B*H). Block: 256 threads (8 waves), each wave owns 16 query rows x d=64.
// Scores arrive pre-scaled by (1/sqrt(d))*log2(e): softmax runs in base-2 so every
// exponential is a single v_exp_f32. Row-sums accumulate via ones-column WMMA.
#define QB 128
#define KT 64

__global__ __launch_bounds__(256) void attn_k(const bf16_t* __restrict__ Q,
                                              const bf16_t* __restrict__ K,
                                              const bf16_t* __restrict__ VT,  // [b,h,d,k]
                                              bf16_t* __restrict__ Z, int L) {
    __shared__ __align__(16) bf16_t Qs[QB][72];
    __shared__ __align__(16) bf16_t Ks[2][KT][72];       // natural [k][d]
    __shared__ __align__(16) bf16_t VTs[2][64][72];      // [d][k]
    __shared__ __align__(16) bf16_t Ps[8][16][72];       // per-wave P scratch

    const int tid  = threadIdx.x;
    const int lane = tid & 31;
    const int wave = tid >> 5;
    const int hi   = lane >> 4;
    const int l16  = lane & 15;
    const int q0   = blockIdx.x * QB;
    const int bh   = blockIdx.y;
    const int bb   = bh >> 3, h = bh & 7;
    const size_t base  = (size_t)bb * L * 512 + (size_t)h * 64;   // Q/K/Z row base
    const size_t vbase = (size_t)bh * 64 * L;                     // VT row base

    auto stageKV = [&](int buf, int k0) {
        {   // K tile (64x64) row-major
            int r = tid >> 2, cb = (tid & 3) * 16;
            const bf16_t* src = K + base + (size_t)(k0 + r) * 512 + cb;
            copy16_async(src,     &Ks[buf][r][cb]);
            copy16_async(src + 8, &Ks[buf][r][cb + 8]);
        }
        {   // V tile already transposed in global: rows are d, contiguous k
            int dd = tid >> 2, kc = (tid & 3) * 16;
            const bf16_t* src = VT + vbase + (size_t)dd * L + k0 + kc;
            copy16_async(src,     &VTs[buf][dd][kc]);
            copy16_async(src + 8, &VTs[buf][dd][kc + 8]);
        }
    };

    // stage Q block (128 x 64) + first K/V tile
    {
        int r = tid >> 1, cb = (tid & 1) * 32;
        const bf16_t* src = Q + base + (size_t)(q0 + r) * 512 + cb;
        #pragma unroll
        for (int i = 0; i < 4; ++i) copy16_async(src + i * 8, &Qs[r][cb + i * 8]);
    }
    stageKV(0, 0);
    async_wait0();
    __syncthreads();

    // Q fragments are loop-invariant: hoist into registers (2 K-chunks of d)
    FragAB faQ[2];
    #pragma unroll
    for (int c = 0; c < 2; ++c) {
        faQ[c].q[0] = *reinterpret_cast<const uint4*>(&Qs[wave * 16 + l16][c * 32 + hi * 8]);
        faQ[c].q[1] = *reinterpret_cast<const uint4*>(&Qs[wave * 16 + l16][c * 32 + 16 + hi * 8]);
    }

    // constant ones B-fragment: column 0 of B is all 1.0 -> accL accumulates row-sums of P
    FragAB fones;
    {
        unsigned ob = (l16 == 0) ? 0x3f803f80u : 0u;
        fones.q[0] = make_uint4(ob, ob, ob, ob);
        fones.q[1] = make_uint4(ob, ob, ob, ob);
    }

    float mrow[8];
    v8f accO[4] = {};
    v8f accL = {};
    #pragma unroll
    for (int i = 0; i < 8; ++i) mrow[i] = -1e30f;

    int buf = 0;
    for (int k0 = 0; k0 < L; k0 += KT, buf ^= 1) {
        if (k0 + KT < L) stageKV(buf ^ 1, k0 + KT);   // prefetch next tile async

        // ---- S = Q * K^T  (already in log2 domain)
        v8f accS[4] = {};
        #pragma unroll
        for (int c = 0; c < 2; ++c) {
            #pragma unroll
            for (int t = 0; t < 4; ++t) {
                FragAB fb;   // Ks[k][d] is exactly B layout: d in table order at key column
                fb.q[0] = *reinterpret_cast<const uint4*>(&Ks[buf][t * 16 + l16][c * 32 + hi * 8]);
                fb.q[1] = *reinterpret_cast<const uint4*>(&Ks[buf][t * 16 + l16][c * 32 + 16 + hi * 8]);
                accS[t] = wmma_bf16(faQ[c], fb, accS[t]);
            }
        }

        // ---- row max: batched step-outer reduction
        float mx[8];
        #pragma unroll
        for (int i = 0; i < 8; ++i)
            mx[i] = fmaxf(fmaxf(accS[0][i], accS[1][i]), fmaxf(accS[2][i], accS[3][i]));
        #pragma unroll
        for (int d = 8; d >= 1; d >>= 1) {
            float o[8];
            #pragma unroll
            for (int i = 0; i < 8; ++i) o[i] = __shfl_xor(mx[i], d, 32);
            #pragma unroll
            for (int i = 0; i < 8; ++i) mx[i] = fmaxf(mx[i], o[i]);
        }

        // ---- online rescale + exponentiate, all base-2 (single v_exp per element)
        #pragma unroll
        for (int i = 0; i < 8; ++i) {
            float mnew = fmaxf(mrow[i], mx[i]);
            float corr = exp2_fast(mrow[i] - mnew);
            mrow[i] = mnew;
            #pragma unroll
            for (int t = 0; t < 4; ++t) accS[t][i] = exp2_fast(accS[t][i] - mnew);
            #pragma unroll
            for (int t = 0; t < 4; ++t) accO[t][i] *= corr;
            accL[i] *= corr;
        }

        // ---- D-layout P -> per-wave LDS -> A-layout fragments
        #pragma unroll
        for (int t = 0; t < 4; ++t)
            #pragma unroll
            for (int i = 0; i < 8; ++i)
                Ps[wave][i + 8 * hi][t * 16 + l16] = (bf16_t)accS[t][i];

        // ---- accO += P * V ; accL += P * ones
        #pragma unroll
        for (int c = 0; c < 2; ++c) {
            FragAB fa;
            fa.q[0] = *reinterpret_cast<const uint4*>(&Ps[wave][l16][c * 32 + hi * 8]);
            fa.q[1] = *reinterpret_cast<const uint4*>(&Ps[wave][l16][c * 32 + 16 + hi * 8]);
            accL = wmma_bf16(fa, fones, accL);
            #pragma unroll
            for (int t = 0; t < 4; ++t) {
                FragAB fb;
                fb.q[0] = *reinterpret_cast<const uint4*>(&VTs[buf][t * 16 + l16][c * 32 + hi * 8]);
                fb.q[1] = *reinterpret_cast<const uint4*>(&VTs[buf][t * 16 + l16][c * 32 + 16 + hi * 8]);
                accO[t] = wmma_bf16(fa, fb, accO[t]);
            }
        }
        async_wait0();
        __syncthreads();
    }

    // ---- epilogue: broadcast row-sums from column 0 of accL, normalize, store
    float rinv[8];
    #pragma unroll
    for (int i = 0; i < 8; ++i) {
        float ls = __shfl(accL[i], hi * 16, 32);    // lane l16==0 of this half holds the sum
        rinv[i] = rcp_fast(ls);
    }
    #pragma unroll
    for (int t = 0; t < 4; ++t)
        #pragma unroll
        for (int i = 0; i < 8; ++i) {
            int row = q0 + wave * 16 + i + 8 * hi;
            Z[base + (size_t)row * 512 + t * 16 + l16] = (bf16_t)(accO[t][i] * rinv[i]);
        }
}

// ---------------------------------------------------------------- driver
extern "C" void kernel_launch(void* const* d_in, const int* in_sizes, int n_in,
                              void* d_out, int out_size, void* d_ws, size_t ws_size,
                              hipStream_t stream) {
    const float* x  = (const float*)d_in[0];
    const float* Wq = (const float*)d_in[1];
    const float* bq = (const float*)d_in[2];
    const float* Wk = (const float*)d_in[3];
    const float* bk = (const float*)d_in[4];
    const float* Wv = (const float*)d_in[5];
    const float* bv = (const float*)d_in[6];
    const float* Wo = (const float*)d_in[7];
    const float* bo = (const float*)d_in[8];

    const int Bsz = 4, L = 2048, D = 512;
    const int M = Bsz * L;                       // 8192

    size_t off = 0;
    auto carve = [&](size_t bytes) {
        void* p = (char*)d_ws + off;
        off += (bytes + 255) & ~(size_t)255;
        return p;
    };
    bf16_t* xb  = (bf16_t*)carve((size_t)M * D * 2);
    bf16_t* wqt = (bf16_t*)carve((size_t)D * D * 2);
    bf16_t* wkt = (bf16_t*)carve((size_t)D * D * 2);
    bf16_t* wvt = (bf16_t*)carve((size_t)D * D * 2);
    bf16_t* wot = (bf16_t*)carve((size_t)D * D * 2);
    bf16_t* qb  = (bf16_t*)carve((size_t)M * D * 2);
    bf16_t* kb  = (bf16_t*)carve((size_t)M * D * 2);
    bf16_t* vtg = (bf16_t*)carve((size_t)M * D * 2);   // [b,h,d,k]
    bf16_t* zb  = (bf16_t*)carve((size_t)M * D * 2);

    // fp32 -> bf16 staging; weights transposed so GEMM B-tiles stage with async b128
    f32_to_bf16_k<<<(M * D + 255) / 256, 256, 0, stream>>>(x, xb, M * D);
    dim3 gt(16, 16);
    transpose_f32_to_bf16_k<<<gt, 256, 0, stream>>>(Wq, wqt);
    transpose_f32_to_bf16_k<<<gt, 256, 0, stream>>>(Wk, wkt);
    transpose_f32_to_bf16_k<<<gt, 256, 0, stream>>>(Wv, wvt);
    transpose_f32_to_bf16_k<<<gt, 256, 0, stream>>>(Wo, wot);

    // projections; Q carries (1/sqrt(64)) * log2(e) so softmax runs in base-2
    const float qscale = 0.125f * 1.4426950408889634f;
    dim3 gg(D / BN, M / BM);                     // (4, 64)
    gemm512_k<1><<<gg, 256, 0, stream>>>(xb, wqt, bq, qb,  nullptr, qscale);
    gemm512_k<1><<<gg, 256, 0, stream>>>(xb, wkt, bk, kb,  nullptr, 1.0f);
    gemm512_k<2><<<gg, 256, 0, stream>>>(xb, wvt, bv, vtg, nullptr, 1.0f);

    // attention
    dim3 ga(L / QB, Bsz * 8);                    // (16, 32)
    attn_k<<<ga, 256, 0, stream>>>(qb, kb, vtg, zb, L);

    // output projection -> fp32 d_out
    gemm512_k<0><<<gg, 256, 0, stream>>>(zb, wot, bo, nullptr, (float*)d_out, 1.0f);
}
